// EmbedAlignerOT_75814762709620
// MI455X (gfx1250) — matrix operand order repaired and checked by reading, not compiled, not run on hardware
//
#include <hip/hip_runtime.h>
#include <math.h>

// ---------- types for WMMA ----------
typedef __bf16 bf16_t;
typedef bf16_t v16bf __attribute__((ext_vector_type(16)));
typedef float  v8f   __attribute__((ext_vector_type(8)));

union FragU { v16bf v; uint4 q[2]; };

__device__ __forceinline__ unsigned short f2bf(float f) {
    unsigned int u = __float_as_uint(f);
    unsigned int r = u + 0x7FFFu + ((u >> 16) & 1u);   // round-to-nearest-even
    return (unsigned short)(r >> 16);
}

// A fragment, 16x32 bf16: lane (M=l&15, hi=l>>4) holds
// K = hi*16 .. hi*16+15 (matches B)  -> two 16B chunks.
__device__ __forceinline__ v16bf load_fragA(const unsigned short* rowk0, int hi) {
    FragU f;
    f.q[0] = *(const uint4*)(rowk0 + hi * 16);
    f.q[1] = *(const uint4*)(rowk0 + hi * 16 + 8);
    return f.v;
}
// B fragment: lane (N=l&15, hi) holds K = hi*16 .. hi*16+15 -> 32B contiguous.
__device__ __forceinline__ v16bf load_fragB_mem(const unsigned short* p) {
    FragU f;
    f.q[0] = *(const uint4*)(p);
    f.q[1] = *(const uint4*)(p + 8);
    return f.v;
}

// ---- gfx1250 async global->LDS copy (ASYNCcnt path, ISA 15.18.3 op 98) ----
// Moves 32B: INST_OFFSET is added to BOTH the LDS and global addresses.
__device__ __forceinline__ void async_cp32(unsigned ldsoff, const unsigned short* g) {
    asm volatile("global_load_async_to_lds_b128 %0, %1, off\n\t"
                 "global_load_async_to_lds_b128 %0, %1, off offset:16"
                 :: "v"(ldsoff), "v"(g) : "memory");
}
__device__ __forceinline__ void wait_async0() {
    asm volatile("s_wait_asynccnt 0" ::: "memory");
}

// ---------- prep: row-normalize -> bf16, plus raw source transposed bf16 ----------
__global__ void prep_kernel(const float* __restrict__ src, const float* __restrict__ tgt,
                            unsigned short* __restrict__ xnbf, unsigned short* __restrict__ ynbf,
                            unsigned short* __restrict__ sTbf, int n, int d) {
    int row = blockIdx.x;
    bool isSrc = row < n;
    int r = isSrc ? row : row - n;
    const float* p = (isSrc ? src : tgt) + (size_t)r * d;

    float vals[4];
    float ss = 0.f;
    #pragma unroll
    for (int q = 0; q < 4; ++q) {
        int c = threadIdx.x + q * 256;
        vals[q] = p[c];
        ss += vals[q] * vals[q];
    }
    __shared__ float red[256];
    red[threadIdx.x] = ss;
    __syncthreads();
    for (int s = 128; s > 0; s >>= 1) {
        if (threadIdx.x < s) red[threadIdx.x] += red[threadIdx.x + s];
        __syncthreads();
    }
    float rn = rsqrtf(red[0]);

    unsigned short* outp = (isSrc ? xnbf : ynbf) + (size_t)r * d;
    #pragma unroll
    for (int q = 0; q < 4; ++q) {
        int c = threadIdx.x + q * 256;
        outp[c] = f2bf(vals[q] * rn);
        if (isSrc) sTbf[(size_t)c * n + r] = f2bf(vals[q]);  // source^T (raw, un-normalized)
    }
}

__global__ void init_uv_kernel(float* __restrict__ u, float* __restrict__ v, int n, float val) {
    int i = blockIdx.x * blockDim.x + threadIdx.x;
    if (i < n) { u[i] = val; v[i] = val; }
}

// ---------- WMMA NT-GEMM: C[i,j] = sum_k A[i,k]*Bcols[j,k] ----------
// Block tile 128x128: 8 waves as 4(M) x 2(N), each wave 32x64 = 2x4 WMMA tiles.
// B tile (128 cols x 32 K, bf16) double-buffered in LDS via async global->LDS.
// EPI==0: Cout = A (f32), Kout = exp(A) (f32)     [GEMM1 + Sinkhorn kernel matrix]
// EPI==1: Cout = result (f32)                     [GEMM2: aligned embeddings]
template <int EPI>
__global__ __launch_bounds__(256, 2)
void gemm_nt_wmma(const unsigned short* __restrict__ Abf,
                  const unsigned short* __restrict__ Bbf,
                  int Kdim, int ldc,
                  float* __restrict__ Cout, float* __restrict__ Kout) {
    constexpr int CSTR = 40;             // shorts per LDS column slot (64B data + 16B pad)
    constexpr int BUFS = 128 * CSTR;     // shorts per buffer (10240 B)
    __shared__ __align__(128) unsigned short Bs[2 * BUFS];

    const int tid  = threadIdx.x;
    const int lane = tid & 31;
    const int w    = tid >> 5;
    const int wrow = w & 3;              // 4 waves along M
    const int wcol = w >> 2;             // 2 waves along N
    const int hi   = lane >> 4;
    const int ln   = lane & 15;

    const int i0   = blockIdx.x * 128 + wrow * 32;
    const int jblk = blockIdx.y * 128;
    const int j0   = jblk + wcol * 64;

    // staging plan: thread t moves 32B of B column (t>>1), K-half (t&1)
    const int scol  = tid >> 1;
    const int shalf = tid & 1;
    const unsigned short* gB = Bbf + (size_t)(jblk + scol) * Kdim + shalf * 16;
    const unsigned ldsW = (unsigned)(uintptr_t)(&Bs[scol * CSTR + shalf * 16]);

    const unsigned short* arow0 = Abf + (size_t)(i0 + ln)      * Kdim;
    const unsigned short* arow1 = Abf + (size_t)(i0 + 16 + ln) * Kdim;

    v8f acc[2][4] = {};

    async_cp32(ldsW, gB);                // prologue: stage k0 = 0 into buffer 0

    int buf = 0;
    for (int k0 = 0; k0 < Kdim; k0 += 32) {
        wait_async0();                   // my async writes for current buffer done
        __syncthreads();                 // publish all waves' writes
        if (k0 + 32 < Kdim) {            // stream next tile while computing
            async_cp32(ldsW + (buf ^ 1) * (BUFS * 2), gB + k0 + 32);
            __builtin_prefetch(arow0 + k0 + 32, 0, 1);   // global_prefetch_b8
        }
        v16bf a0 = load_fragA(arow0 + k0, hi);
        v16bf a1 = load_fragA(arow1 + k0, hi);
        #pragma unroll
        for (int ni = 0; ni < 4; ++ni) {
            const unsigned short* bp =
                &Bs[buf * BUFS + (wcol * 64 + ni * 16 + ln) * CSTR + hi * 16];
            v16bf b = load_fragB_mem(bp);
            acc[0][ni] = __builtin_amdgcn_wmma_f32_16x16x32_bf16(false, a0, false, b, (short)0, acc[0][ni], false, false);
            acc[1][ni] = __builtin_amdgcn_wmma_f32_16x16x32_bf16(false, a1, false, b, (short)0, acc[1][ni], false, false);
        }
        buf ^= 1;
    }

    // C/D layout (7.12.2): VGPR r, lane l -> row = r + 8*hi, col = l&15
    #pragma unroll
    for (int mi = 0; mi < 2; ++mi) {
        #pragma unroll
        for (int ni = 0; ni < 4; ++ni) {
            int jcol = j0 + ni * 16 + ln;
            #pragma unroll
            for (int r = 0; r < 8; ++r) {
                int irow = i0 + mi * 16 + hi * 8 + r;
                float val = acc[mi][ni][r];
                size_t idx = (size_t)irow * ldc + jcol;
                if (EPI == 0) {
                    Cout[idx] = val;            // A
                    Kout[idx] = __expf(val);    // K = exp(A)
                } else {
                    Cout[idx] = val;            // aligned embeddings
                }
            }
        }
    }
}

// ---------- Sinkhorn half-iterations (K f32 = 64MB, resident in 192MB L2) ----------
// u[i] = sqrt(scale / sum_j K[i,j] * v[j])  : one block per row, coalesced row read
__global__ void sk_row_update(const float* __restrict__ K, const float* __restrict__ v,
                              float* __restrict__ u, int m, float scale) {
    int i = blockIdx.x;
    const float* row = K + (size_t)i * m;
    float acc = 0.f;
    for (int j = threadIdx.x; j < m; j += 256)
        acc += row[j] * v[j];
    __shared__ float red[256];
    red[threadIdx.x] = acc;
    __syncthreads();
    for (int s = 128; s > 0; s >>= 1) {
        if (threadIdx.x < s) red[threadIdx.x] += red[threadIdx.x + s];
        __syncthreads();
    }
    if (threadIdx.x == 0) u[i] = sqrtf(scale / red[0]);
}

// v[j] = sqrt(scale / sum_i K[i,j] * u[i])  : thread per column, coalesced across lanes
__global__ void sk_col_update(const float* __restrict__ K, const float* __restrict__ u,
                              float* __restrict__ v, int n, int m, float scale) {
    int j = blockIdx.x * blockDim.x + threadIdx.x;
    float acc = 0.f;
    const float* p = K + j;
    for (int i = 0; i < n; i += 4) {
        acc += p[(size_t)i       * m] * u[i]
             + p[(size_t)(i + 1) * m] * u[i + 1]
             + p[(size_t)(i + 2) * m] * u[i + 2]
             + p[(size_t)(i + 3) * m] * u[i + 3];
    }
    v[j] = sqrtf(scale / acc);
}

// ---------- T = A * (u K v); also emit T^T as bf16 for GEMM2 ----------
__global__ void t_kernel(const float* __restrict__ A, const float* __restrict__ K,
                         const float* __restrict__ u, const float* __restrict__ v,
                         float* __restrict__ Tout, unsigned short* __restrict__ Ttbf,
                         int n, int m) {
    int j = blockIdx.x * 256 + threadIdx.x;
    int i = blockIdx.y;
    size_t idx = (size_t)i * m + j;
    float t = A[idx] * (u[i] * K[idx] * v[j]);
    Tout[idx] = t;
    Ttbf[(size_t)j * n + i] = f2bf(t);   // scattered 2B writes, but T^T (32MB) is L2-resident
}

extern "C" void kernel_launch(void* const* d_in, const int* in_sizes, int n_in,
                              void* d_out, int out_size, void* d_ws, size_t ws_size,
                              hipStream_t stream) {
    const int d = 1024;
    const int n = in_sizes[0] / d;   // 4096
    const int m = in_sizes[1] / d;   // 4096

    const float* src = (const float*)d_in[0];
    const float* tgt = (const float*)d_in[1];

    float* aligned = (float*)d_out;                        // [m, d]
    float* Tout    = (float*)d_out + (size_t)m * d;        // [n, m]

    char* w = (char*)d_ws;
    unsigned short* xnbf = (unsigned short*)(w);                                   // [n,d] bf16
    unsigned short* ynbf = (unsigned short*)(w + (size_t)n * d * 2);               // [m,d] bf16
    unsigned short* sTbf = (unsigned short*)(w + (size_t)(n + m) * d * 2);         // [d,n] bf16
    unsigned short* Ttbf = (unsigned short*)(w + (size_t)(n + m + n) * d * 2);     // [m,n] bf16
    char* w2 = (char*)(Ttbf) + (size_t)m * n * 2;
    float* Afull = (float*)(w2);                                                   // [n,m] f32
    float* Kf    = (float*)(w2 + (size_t)n * m * 4);                               // [n,m] f32
    float* uu    = (float*)(w2 + (size_t)n * m * 8);
    float* vv    = uu + n;

    // 1) normalize + bf16 conversions (+ source^T)
    prep_kernel<<<dim3(n + m), 256, 0, stream>>>(src, tgt, xnbf, ynbf, sTbf, n, d);
    init_uv_kernel<<<dim3((n + 255) / 256), 256, 0, stream>>>(uu, vv, n, 1.0f / (float)n);

    // 2) GEMM1: A = xn @ yn^T (bf16 WMMA, f32 acc); epilogue stores A and K=exp(A)
    gemm_nt_wmma<0><<<dim3(n / 128, m / 128), 256, 0, stream>>>(xnbf, ynbf, d, m, Afull, Kf);

    // 3) 200 unbalanced-Sinkhorn iterations (fi = reg_m/(reg_m+reg) = 0.5)
    const float scale = 1.0f / (float)n;   // a_i = b_j = 1/n
    for (int it = 0; it < 200; ++it) {
        sk_row_update<<<dim3(n), 256, 0, stream>>>(Kf, vv, uu, m, scale);
        sk_col_update<<<dim3(m / 256), 256, 0, stream>>>(Kf, uu, vv, n, m, scale);
    }

    // 4) T = A * u K v ; emit T^T bf16 for GEMM2
    t_kernel<<<dim3(m / 256, n), 256, 0, stream>>>(Afull, Kf, uu, vv, Tout, Ttbf, n, m);

    // 5) GEMM2: aligned = T^T @ source  (A = T^T [m,n] bf16, Bcols = source^T [d,n] bf16)
    gemm_nt_wmma<1><<<dim3(m / 128, d / 128), 256, 0, stream>>>(Ttbf, sTbf, n, d, aligned, nullptr);
}
